// MultiHeadAttention_43319040147969
// MI455X (gfx1250) — matrix operand order repaired and checked by reading, compiled, not run on hardware
//
#include <hip/hip_runtime.h>

// ---------------------------------------------------------------------------
// MHA for MI455X (gfx1250, wave32).  All matmuls via v_wmma_f32_16x16x32_bf16.
// Workspace layout (needs 32 MB):
//   [ 0MB) qh  bf16 [B,H,S,Dh]
//   [ 8MB) kh  bf16 [B,H,S,Dh]
//   [16MB) vhT bf16 [B,H,Dh,S]   (transposed so PV B-fragments are contiguous)
//   [24MB) ctx bf16 [B,S,D]
// ---------------------------------------------------------------------------

typedef unsigned short u16;
typedef unsigned int   u32;
typedef __attribute__((ext_vector_type(16))) __bf16 v16bf;
typedef __attribute__((ext_vector_type(8)))  float  v8f;

union Frag {
  v16bf v;
  uint4 q[2];
};

__device__ __forceinline__ u16 f2bf(float f) {        // round-to-nearest-even
  u32 u = __float_as_uint(f);
  u32 r = (u + 0x7FFFu + ((u >> 16) & 1u)) >> 16;
  return (u16)r;
}
__device__ __forceinline__ u32 pack2(float a, float b) {
  return (u32)f2bf(a) | ((u32)f2bf(b) << 16);
}
__device__ __forceinline__ v8f wmma_bf16(const Frag& a, const Frag& b, v8f c) {
  return __builtin_amdgcn_wmma_f32_16x16x32_bf16(
      /*neg_a=*/false, a.v, /*neg_b=*/false, b.v,
      /*c_mod=*/(short)0, c, /*reuse_a=*/false, /*reuse_b=*/false);
}

// ---------------------------------------------------------------------------
// GEMM: Y = A(M x K) * W(N x K)^T + bias.   M=4096, N=K=1024.
// One wave computes a 32x64 macro-tile: 2 A-frags x 4 B-frags -> 8 WMMAs per
// 32-wide k-step (each staged fragment reused 2-4x).  8 waves/block.
// MODE 0: bf16 out, head-major [B,H,S,Dh]
// MODE 1: bf16 out, transposed [B,H,Dh,S]
// MODE 2: f32  out, row-major  [M,N]
// ---------------------------------------------------------------------------
template <int MODE, bool ABF16>
__global__ __launch_bounds__(256) void gemm16(const void* __restrict__ Ap,
                                              const float* __restrict__ W,
                                              const float* __restrict__ bias,
                                              void* __restrict__ Yp) {
  constexpr int N = 1024, K = 1024;
  __shared__ __align__(16) u16 ldsA[8][32 * 32];   // 16 KB
  __shared__ __align__(16) u16 ldsB[8][64 * 32];   // 32 KB

  const int w    = threadIdx.x >> 5;
  const int lane = threadIdx.x & 31;
  const int row  = lane & 15;
  const int hlf  = lane >> 4;
  const int tile = blockIdx.x * 8 + w;      // 0..2047
  const int tM   = (tile >> 4) << 5;        // tile / (N/64) * 32
  const int tN   = (tile & 15) << 6;        // (tile % 16) * 64

  const int kb  = hlf * 8;    // A-fragment K base
  const int kb2 = hlf * 8;    // B-fragment K base

  v8f acc[2][4] = {};
  for (int k0 = 0; k0 < K; k0 += 32) {
    // ---- stage A macro-tile (32 rows x 32 k) as bf16 in LDS
#pragma unroll
    for (int r = 0; r < 2; ++r) {
      const int ar = r * 16 + row;
      u16* dst = &ldsA[w][ar * 32 + hlf * 16];
      if constexpr (ABF16) {
        const u16* src = (const u16*)Ap + (size_t)(tM + ar) * K + k0 + hlf * 16;
        if (k0 + 32 < K) __builtin_prefetch(src + 32, 0, 1);
        ((uint4*)dst)[0] = ((const uint4*)src)[0];
        ((uint4*)dst)[1] = ((const uint4*)src)[1];
      } else {
        const float* src = (const float*)Ap + (size_t)(tM + ar) * K + k0 + hlf * 16;
        if (k0 + 32 < K) __builtin_prefetch(src + 32, 0, 1);
        float4 f0 = ((const float4*)src)[0];
        float4 f1 = ((const float4*)src)[1];
        float4 f2 = ((const float4*)src)[2];
        float4 f3 = ((const float4*)src)[3];
        uint4 p0 = {pack2(f0.x, f0.y), pack2(f0.z, f0.w), pack2(f1.x, f1.y), pack2(f1.z, f1.w)};
        uint4 p1 = {pack2(f2.x, f2.y), pack2(f2.z, f2.w), pack2(f3.x, f3.y), pack2(f3.z, f3.w)};
        ((uint4*)dst)[0] = p0;
        ((uint4*)dst)[1] = p1;
      }
    }
    // ---- stage B macro-tile = 64 W rows x 32 k as bf16 in LDS
#pragma unroll
    for (int r = 0; r < 4; ++r) {
      const int br = r * 16 + row;
      const float* src = W + (size_t)(tN + br) * K + k0 + hlf * 16;
      if (k0 + 32 < K) __builtin_prefetch(src + 32, 0, 1);
      float4 f0 = ((const float4*)src)[0];
      float4 f1 = ((const float4*)src)[1];
      float4 f2 = ((const float4*)src)[2];
      float4 f3 = ((const float4*)src)[3];
      uint4 p0 = {pack2(f0.x, f0.y), pack2(f0.z, f0.w), pack2(f1.x, f1.y), pack2(f1.z, f1.w)};
      uint4 p1 = {pack2(f2.x, f2.y), pack2(f2.z, f2.w), pack2(f3.x, f3.y), pack2(f3.z, f3.w)};
      u16* dst = &ldsB[w][br * 32 + hlf * 16];
      ((uint4*)dst)[0] = p0;
      ((uint4*)dst)[1] = p1;
    }
    __syncthreads();

    Frag a0, a1, b[4];
    a0.q[0] = *(const uint4*)&ldsA[w][row * 32 + kb];
    a0.q[1] = *(const uint4*)&ldsA[w][row * 32 + 16 + kb];
    a1.q[0] = *(const uint4*)&ldsA[w][(16 + row) * 32 + kb];
    a1.q[1] = *(const uint4*)&ldsA[w][(16 + row) * 32 + 16 + kb];
#pragma unroll
    for (int ni = 0; ni < 4; ++ni) {
      b[ni].q[0] = *(const uint4*)&ldsB[w][(ni * 16 + row) * 32 + kb2];
      b[ni].q[1] = *(const uint4*)&ldsB[w][(ni * 16 + row) * 32 + kb2 + 16];
    }
#pragma unroll
    for (int ni = 0; ni < 4; ++ni) {
      acc[0][ni] = wmma_bf16(a0, b[ni], acc[0][ni]);
      acc[1][ni] = wmma_bf16(a1, b[ni], acc[1][ni]);
    }
    __syncthreads();
  }

  // ---- epilogue: bias + store (C column for every acc element is lane&15)
#pragma unroll
  for (int ni = 0; ni < 4; ++ni) {
    const float bval = bias[tN + ni * 16 + row];
    const int gn = tN + ni * 16 + row;
#pragma unroll
    for (int mi = 0; mi < 2; ++mi) {
#pragma unroll
      for (int i = 0; i < 8; ++i) {
        float c = acc[mi][ni][i] + bval;
        int gm = tM + mi * 16 + i + hlf * 8;
        if constexpr (MODE == 2) {
          ((float*)Yp)[(size_t)gm * N + gn] = c;
        } else {
          int bb = gm >> 11, ss = gm & 2047;   // batch / seq
          int hh = gn >> 6,  dd = gn & 63;     // head / head-dim
          u16* Y = (u16*)Yp;
          if constexpr (MODE == 0)
            Y[(((size_t)bb * 16 + hh) * 2048 + ss) * 64 + dd] = f2bf(c);
          else
            Y[(((size_t)bb * 16 + hh) * 64 + dd) * 2048 + ss] = f2bf(c);
        }
      }
    }
  }
}

// ---------------------------------------------------------------------------
// Flash attention: one wave per 16-query tile of one (b,h); streams 32-key
// tiles.  QK^T: 4 WMMAs/tile, PV: 4 WMMAs/tile, online softmax in between.
// ---------------------------------------------------------------------------
__global__ __launch_bounds__(256) void attn16(const u16* __restrict__ qh,
                                              const u16* __restrict__ kh,
                                              const u16* __restrict__ vhT,
                                              u16* __restrict__ ctx) {
  __shared__ __align__(16) u16 ldsP[8][16 * 32];

  const int w    = threadIdx.x >> 5;
  const int lane = threadIdx.x & 31;
  const int row  = lane & 15;
  const int hlf  = lane >> 4;
  const int gw   = blockIdx.x * 8 + w;   // 0..4095
  const int qt   = gw & 127;             // query tile within (b,h)
  const int bh   = gw >> 7;              // fused (b,h) 0..31
  const int hh   = bh & 15;
  const int bb   = bh >> 4;

  const u16* qbase = qh + (((size_t)bh) * 2048 + qt * 16) * 64;
  const u16* kbase = kh + ((size_t)bh) * 2048 * 64;
  const u16* vbase = vhT + ((size_t)bh) * 64 * 2048;

  const int kb  = hlf * 16;
  const int kb2 = hlf * 16;

  // Q fragments (loop invariant): head-dims 0..31 and 32..63
  Frag a0, a1;
  {
    const u16* qr = qbase + row * 64;
    a0.q[0] = *(const uint4*)(qr + kb);
    a0.q[1] = *(const uint4*)(qr + kb + 8);
    a1.q[0] = *(const uint4*)(qr + 32 + kb);
    a1.q[1] = *(const uint4*)(qr + 40 + kb);
  }

  float mrow[8], lrow[8];
  v8f o0 = {}, o1 = {}, o2 = {}, o3 = {};
#pragma unroll
  for (int i = 0; i < 8; ++i) { mrow[i] = -3.0e38f; lrow[i] = 0.0f; }

  for (int key0 = 0; key0 < 2048; key0 += 32) {
    // ---- scores = Q * K^T for 32 keys (two 16-key column tiles)
    Frag bk0a, bk0b, bk1a, bk1b;
    {
      const u16* kr0 = kbase + (size_t)(key0 + row) * 64;
      bk0a.q[0] = *(const uint4*)(kr0 + kb2);
      bk0a.q[1] = *(const uint4*)(kr0 + kb2 + 8);
      bk0b.q[0] = *(const uint4*)(kr0 + 32 + kb2);
      bk0b.q[1] = *(const uint4*)(kr0 + 32 + kb2 + 8);
      const u16* kr1 = kbase + (size_t)(key0 + 16 + row) * 64;
      bk1a.q[0] = *(const uint4*)(kr1 + kb2);
      bk1a.q[1] = *(const uint4*)(kr1 + kb2 + 8);
      bk1b.q[0] = *(const uint4*)(kr1 + 32 + kb2);
      bk1b.q[1] = *(const uint4*)(kr1 + 32 + kb2 + 8);
    }
    v8f sc0 = {}, sc1 = {};
    sc0 = wmma_bf16(a0, bk0a, sc0);
    sc0 = wmma_bf16(a1, bk0b, sc0);
    sc1 = wmma_bf16(a0, bk1a, sc1);
    sc1 = wmma_bf16(a1, bk1b, sc1);

    // ---- online softmax (rows live across 16-lane halves -> xor butterflies)
    float p0[8], p1[8], scl[8];
#pragma unroll
    for (int i = 0; i < 8; ++i) {
      float s0 = sc0[i] * 0.125f;   // 1/sqrt(64)
      float s1 = sc1[i] * 0.125f;
      float t = fmaxf(s0, s1);
      t = fmaxf(t, __shfl_xor(t, 1, 32));
      t = fmaxf(t, __shfl_xor(t, 2, 32));
      t = fmaxf(t, __shfl_xor(t, 4, 32));
      t = fmaxf(t, __shfl_xor(t, 8, 32));
      float mn = fmaxf(mrow[i], t);
      scl[i] = __expf(mrow[i] - mn);
      mrow[i] = mn;
      p0[i] = __expf(s0 - mn);
      p1[i] = __expf(s1 - mn);
      float rs = p0[i] + p1[i];
      rs += __shfl_xor(rs, 1, 32);
      rs += __shfl_xor(rs, 2, 32);
      rs += __shfl_xor(rs, 4, 32);
      rs += __shfl_xor(rs, 8, 32);
      lrow[i] = lrow[i] * scl[i] + rs;
    }
#pragma unroll
    for (int i = 0; i < 8; ++i) {
      o0[i] *= scl[i]; o1[i] *= scl[i]; o2[i] *= scl[i]; o3[i] *= scl[i];
    }

    // ---- P: C-layout -> A-fragment layout via LDS round trip (bf16)
#pragma unroll
    for (int i = 0; i < 8; ++i) {
      int m = i + hlf * 8;
      ldsP[w][m * 32 + row]      = f2bf(p0[i]);
      ldsP[w][m * 32 + 16 + row] = f2bf(p1[i]);
    }
    __syncthreads();
    Frag pa;
    pa.q[0] = *(const uint4*)&ldsP[w][row * 32 + kb];
    pa.q[1] = *(const uint4*)&ldsP[w][row * 32 + kb + 8];

    // ---- V fragments from transposed layout: contiguous along key dim
    Frag v0f, v1f, v2f, v3f;
    const u16* vr = vbase + (size_t)row * 2048 + key0 + kb2;
    v0f.q[0] = *(const uint4*)(vr);
    v0f.q[1] = *(const uint4*)(vr + 8);
    v1f.q[0] = *(const uint4*)(vr + 16 * 2048);
    v1f.q[1] = *(const uint4*)(vr + 16 * 2048 + 8);
    v2f.q[0] = *(const uint4*)(vr + 32 * 2048);
    v2f.q[1] = *(const uint4*)(vr + 32 * 2048 + 8);
    v3f.q[0] = *(const uint4*)(vr + 48 * 2048);
    v3f.q[1] = *(const uint4*)(vr + 48 * 2048 + 8);

    o0 = wmma_bf16(pa, v0f, o0);
    o1 = wmma_bf16(pa, v1f, o1);
    o2 = wmma_bf16(pa, v2f, o2);
    o3 = wmma_bf16(pa, v3f, o3);
    __syncthreads();
  }

  // ---- finalize: divide by row sums, merge heads into ctx [B,S,D] (bf16)
  u16* cb = ctx + ((size_t)bb * 2048 + qt * 16) * 1024 + hh * 64;
#pragma unroll
  for (int i = 0; i < 8; ++i) {
    float inv = 1.0f / lrow[i];
    int m = i + hlf * 8;
    u16* cr = cb + (size_t)m * 1024 + row;
    cr[0]  = f2bf(o0[i] * inv);
    cr[16] = f2bf(o1[i] * inv);
    cr[32] = f2bf(o2[i] * inv);
    cr[48] = f2bf(o3[i] * inv);
  }
}

// ---------------------------------------------------------------------------
extern "C" void kernel_launch(void* const* d_in, const int* in_sizes, int n_in,
                              void* d_out, int out_size, void* d_ws, size_t ws_size,
                              hipStream_t stream) {
  const float* q  = (const float*)d_in[0];
  const float* k  = (const float*)d_in[1];
  const float* v  = (const float*)d_in[2];
  const float* wq = (const float*)d_in[3];
  const float* bq = (const float*)d_in[4];
  const float* wk = (const float*)d_in[5];
  const float* bk = (const float*)d_in[6];
  const float* wv = (const float*)d_in[7];
  const float* bv = (const float*)d_in[8];
  const float* wo = (const float*)d_in[9];
  const float* bo = (const float*)d_in[10];

  char* ws  = (char*)d_ws;
  u16*  qh  = (u16*)(ws);
  u16*  kh  = (u16*)(ws + (size_t)8  * 1024 * 1024);
  u16*  vhT = (u16*)(ws + (size_t)16 * 1024 * 1024);
  u16*  ctx = (u16*)(ws + (size_t)24 * 1024 * 1024);

  dim3 blk(256);
  // 4096x1024 output in 32x64 macro-tiles = 2048 tiles, 8 waves per block.
  gemm16<0, false><<<256, blk, 0, stream>>>(q, wq, bq, qh);
  gemm16<0, false><<<256, blk, 0, stream>>>(k, wk, bk, kh);
  gemm16<1, false><<<256, blk, 0, stream>>>(v, wv, bv, vhT);
  // 2*16*128 query tiles, 8 waves per block.
  attn16<<<512, blk, 0, stream>>>(qh, kh, vhT, ctx);
  gemm16<2, true><<<256, blk, 0, stream>>>(ctx, wo, bo, (float*)d_out);
}
